// GCGRUCell_24008867184891
// MI455X (gfx1250) — matrix operand order repaired and checked by reading, not compiled, not run on hardware
//
#include <hip/hip_runtime.h>
#include <math.h>
#include <stdint.h>

// GCGRU cell for MI455X (gfx1250).
// Dominant diffusion GEMMs (~164 GFLOP) run on v_wmma_f32_16x16x32_bf16 with
// hi/lo bf16 operand splitting (3 products, ~2^-17 rel error, f32 accumulate),
// fed by double-buffered GLOBAL_LOAD_ASYNC_TO_LDS_B128 (ASYNCcnt) staging.
// Gate GEMMs (~7 GFLOP) run on v_wmma_f32_16x16x4_f32.
// If ws_size is too small for the bf16 panels, falls back (deterministically)
// to the pure f32-WMMA diffusion path.

#define NN     4000
#define LDP    36    // f32-path LDS pitch (floats)
#define BPITCH 20    // bf16-path LDS pitch (dwords) for 16-dword (32 bf16) rows

typedef __attribute__((ext_vector_type(2))) float    v2f;
typedef __attribute__((ext_vector_type(8))) float    v8f;
typedef __bf16    v16bf  __attribute__((ext_vector_type(16)));
typedef uint32_t  u32x4  __attribute__((ext_vector_type(4)));
typedef uint32_t  u32x2  __attribute__((ext_vector_type(2)));

union BFrag { u32x4 q[2]; v16bf v; };   // all-trivial union for fragment punning

__device__ __forceinline__ v8f wmma4(v2f a, v2f b, v8f c) {
#if defined(__HIP_DEVICE_COMPILE__)
  return __builtin_amdgcn_wmma_f32_16x16x4_f32(false, a, false, b, (short)0, c,
                                               false, false);
#else
  (void)a; (void)b; return c;
#endif
}

__device__ __forceinline__ v8f wmma_bf(v16bf a, v16bf b, v8f c) {
#if defined(__HIP_DEVICE_COMPILE__)
  return __builtin_amdgcn_wmma_f32_16x16x32_bf16(false, a, false, b, (short)0, c,
                                                 false, false);
#else
  (void)a; (void)b; return c;
#endif
}

// async global -> LDS copy, 16B per lane, tracked by ASYNCcnt (no data VGPRs)
__device__ __forceinline__ void async_b128(uint32_t lds_off, const uint32_t* g) {
#if defined(__HIP_DEVICE_COMPILE__)
  asm volatile("global_load_async_to_lds_b128 %0, %1, off"
               :: "v"(lds_off), "v"((uint64_t)(uintptr_t)g)
               : "memory");
#else
  (void)lds_off; (void)g;
#endif
}
__device__ __forceinline__ void wait_async0() {
#if defined(__HIP_DEVICE_COMPILE__)
  asm volatile("s_wait_asynccnt 0" ::: "memory");
#endif
}

__device__ __forceinline__ float sigmoidf_(float v) {
  return 1.0f / (1.0f + expf(-v));
}

// round-to-nearest-even f32 -> bf16 (top 16 bits)
__device__ __forceinline__ uint32_t bf16h(float f) {
  const uint32_t u = __float_as_uint(f);
  return (u + 0x7FFFu + ((u >> 16) & 1u)) >> 16;
}
__device__ __forceinline__ float bf16f(uint32_t hbits) {
  return __uint_as_float(hbits << 16);
}
// split two floats into packed-hi dword and packed-lo dword
struct HLPair { uint32_t H, L; };
__device__ __forceinline__ HLPair split2(float a, float b) {
  const uint32_t h0 = bf16h(a), h1 = bf16h(b);
  const uint32_t l0 = bf16h(a - bf16f(h0)), l1 = bf16h(b - bf16f(h1));
  HLPair p;
  p.H = h0 | (h1 << 16);
  p.L = l0 | (l1 << 16);
  return p;
}

// ---------------------------------------------------------------------------
// pack: Zg[b*96+c][n] = (c<32) ? x[b][c][n] : h[b][c-32][n]; optional hi/lo.
// ---------------------------------------------------------------------------
__global__ __launch_bounds__(256)
void pack_kernel(const float* __restrict__ x, const float* __restrict__ h,
                 float* __restrict__ Zg,
                 uint16_t* __restrict__ ZgH, uint16_t* __restrict__ ZgL) {
  const size_t e = ((size_t)blockIdx.x * blockDim.x + threadIdx.x) * 4;
  const int r = (int)(e / NN);
  const int n = (int)(e - (size_t)r * NN);
  const int b = r / 96;
  const int c = r - b * 96;
  const float* src = (c < 32) ? (x + (size_t)(b * 32 + c) * NN + n)
                              : (h + (size_t)(b * 64 + (c - 32)) * NN + n);
  const float4 v = *(const float4*)src;
  *(float4*)(Zg + e) = v;
  if (ZgH) {
    const HLPair p0 = split2(v.x, v.y);
    const HLPair p1 = split2(v.z, v.w);
    u32x2 Hd, Ld;
    Hd.x = p0.H; Hd.y = p1.H;
    Ld.x = p0.L; Ld.y = p1.L;
    *(u32x2*)((uint32_t*)ZgH + e / 2) = Hd;
    *(u32x2*)((uint32_t*)ZgL + e / 2) = Ld;
  }
}

// ---------------------------------------------------------------------------
// adjsplit: adj(f32) -> AH/AL bf16 panels (row-major [4000][4000])
// ---------------------------------------------------------------------------
__global__ __launch_bounds__(256)
void adjsplit_kernel(const float* __restrict__ A, uint16_t* __restrict__ AH,
                     uint16_t* __restrict__ AL) {
  const size_t e = ((size_t)blockIdx.x * 256 + threadIdx.x) * 8;
  if (e >= (size_t)NN * NN) return;
  const float4 f0 = *(const float4*)(A + e);
  const float4 f1 = *(const float4*)(A + e + 4);
  const HLPair p0 = split2(f0.x, f0.y);
  const HLPair p1 = split2(f0.z, f0.w);
  const HLPair p2 = split2(f1.x, f1.y);
  const HLPair p3 = split2(f1.z, f1.w);
  u32x4 H, L;
  H.x = p0.H; H.y = p1.H; H.z = p2.H; H.w = p3.H;
  L.x = p0.L; L.y = p1.L; L.z = p2.L; L.w = p3.L;
  *(u32x4*)((uint32_t*)AH + e / 2) = H;
  *(u32x4*)((uint32_t*)AL + e / 2) = L;
}

// ---------------------------------------------------------------------------
// bf16 split diffusion: Out[r][m] = sum_n Z[r][n]*adj[m][n]
// Z,adj given as hi/lo bf16 panels. grid = (N/32, R/128), block 128 (4 waves).
// Wave w: 32 rows x 32 m -> 4 f32 accs; 12 bf16 WMMAs per K=32 chunk.
// Staging: double-buffered async global->LDS (next chunk overlaps compute).
// One flat LDS array + constexpr offsets (no LDS addresses in aggregates).
// ---------------------------------------------------------------------------
__device__ __forceinline__ void loadA(BFrag& f, const uint32_t* lds, int p) {
  f.q[0] = *(const u32x4*)(lds + p);      // K 0-7   (this half-lane's view)
  f.q[1] = *(const u32x4*)(lds + p + 8);  // K 16-23
}
__device__ __forceinline__ void loadB(BFrag& f, const uint32_t* lds, int p) {
  f.q[0] = *(const u32x4*)(lds + p);      // K 0-7   (this half-lane's view)
  f.q[1] = *(const u32x4*)(lds + p + 8);  // K 16-23
}

// flat LDS layout (dword offsets)
#define ZSZ (128 * BPITCH)                 // 2560 dwords per Z region
#define BSZ (32 * BPITCH)                  // 640 dwords per B region
#define ZH0_OFF 0
#define ZH1_OFF (ZSZ)
#define ZL0_OFF (2 * ZSZ)
#define ZL1_OFF (3 * ZSZ)
#define BH0_OFF (4 * ZSZ)
#define BH1_OFF (4 * ZSZ + BSZ)
#define BL0_OFF (4 * ZSZ + 2 * BSZ)
#define BL1_OFF (4 * ZSZ + 3 * BSZ)
#define LDS_TOT (4 * ZSZ + 4 * BSZ)        // 12800 dwords = 51.2 KB

__global__ __launch_bounds__(128)
void diffuse_bf16_kernel(const uint16_t* __restrict__ ZH16,
                         const uint16_t* __restrict__ ZL16,
                         const uint16_t* __restrict__ AH16,
                         const uint16_t* __restrict__ AL16,
                         float* __restrict__ Out,
                         uint16_t* __restrict__ OutH,
                         uint16_t* __restrict__ OutL) {
  __shared__ uint32_t lds[LDS_TOT];

  const uint32_t* ZH = (const uint32_t*)ZH16;
  const uint32_t* ZL = (const uint32_t*)ZL16;
  const uint32_t* AH = (const uint32_t*)AH16;
  const uint32_t* AL = (const uint32_t*)AL16;

  const int tid  = threadIdx.x;
  const int wave = tid >> 5;
  const int lane = tid & 31;
  const int l16  = lane & 15;
  const int lh   = lane >> 4;
  const int m0   = blockIdx.x * 32;
  const int r0   = blockIdx.y * 128;

  v8f acc00 = {}, acc01 = {}, acc10 = {}, acc11 = {};

  // staging roles: Z: one 16-dword row-chunk per thread; B: 32 rows x 4 thr
  const uint32_t* zh = ZH + (size_t)(r0 + tid) * (NN / 2);
  const uint32_t* zl = ZL + (size_t)(r0 + tid) * (NN / 2);
  const int br = tid >> 2;
  const int bo = (tid & 3) * 4;
  const uint32_t* ah = AH + (size_t)(m0 + br) * (NN / 2) + bo;
  const uint32_t* al = AL + (size_t)(m0 + br) * (NN / 2) + bo;

  // runtime LDS byte base (addr[31:0] = LDS offset); no aggregates of LDS ptrs
  const uint32_t ldsBase = (uint32_t)(uintptr_t)&lds[0];
  const int zSlot = tid * BPITCH;
  const int bSlot = br * BPITCH + bo;

  auto stage = [&](int p, int k0) {
    const uint32_t zhOff = ldsBase + 4u * (uint32_t)((p ? ZH1_OFF : ZH0_OFF) + zSlot);
    const uint32_t zlOff = ldsBase + 4u * (uint32_t)((p ? ZL1_OFF : ZL0_OFF) + zSlot);
    const uint32_t bhOff = ldsBase + 4u * (uint32_t)((p ? BH1_OFF : BH0_OFF) + bSlot);
    const uint32_t blOff = ldsBase + 4u * (uint32_t)((p ? BL1_OFF : BL0_OFF) + bSlot);
#pragma unroll
    for (int i = 0; i < 4; ++i) {
      async_b128(zhOff + 16 * i, zh + k0 + 4 * i);
      async_b128(zlOff + 16 * i, zl + k0 + 4 * i);
    }
    async_b128(bhOff, ah + k0);
    async_b128(blOff, al + k0);
  };

  const int pa0 = (wave * 32 + l16) * BPITCH + 4 * lh;
  const int pa1 = (wave * 32 + 16 + l16) * BPITCH + 4 * lh;
  const int pb0 = l16 * BPITCH + 4 * lh;
  const int pb1 = (16 + l16) * BPITCH + 4 * lh;

  const int NCHUNK = (NN / 2) / 16;   // 125 chunks of K=32
  stage(0, 0);
  wait_async0();
  __syncthreads();

  for (int c = 0; c < NCHUNK; ++c) {
    const int buf = c & 1;
    if (c + 1 < NCHUNK) stage(buf ^ 1, (c + 1) * 16);  // overlap DMA w/ WMMA

    const uint32_t* ZHp = lds + (buf ? ZH1_OFF : ZH0_OFF);
    const uint32_t* ZLp = lds + (buf ? ZL1_OFF : ZL0_OFF);
    const uint32_t* BHp = lds + (buf ? BH1_OFF : BH0_OFF);
    const uint32_t* BLp = lds + (buf ? BL1_OFF : BL0_OFF);

    // group 1: hi x hi (aH,bH stay live for groups 2/3)
    BFrag aH0, aH1, bH0, bH1;
    loadA(aH0, ZHp, pa0);  loadA(aH1, ZHp, pa1);
    loadB(bH0, BHp, pb0);  loadB(bH1, BHp, pb1);
    acc00 = wmma_bf(aH0.v, bH0.v, acc00);
    acc01 = wmma_bf(aH0.v, bH1.v, acc01);
    acc10 = wmma_bf(aH1.v, bH0.v, acc10);
    acc11 = wmma_bf(aH1.v, bH1.v, acc11);
    // group 2: hi x lo (bL short-lived)
    {
      BFrag bL0, bL1;
      loadB(bL0, BLp, pb0);  loadB(bL1, BLp, pb1);
      acc00 = wmma_bf(aH0.v, bL0.v, acc00);
      acc01 = wmma_bf(aH0.v, bL1.v, acc01);
      acc10 = wmma_bf(aH1.v, bL0.v, acc10);
      acc11 = wmma_bf(aH1.v, bL1.v, acc11);
    }
    // group 3: lo x hi (aL short-lived)
    {
      BFrag aL0, aL1;
      loadA(aL0, ZLp, pa0);  loadA(aL1, ZLp, pa1);
      acc00 = wmma_bf(aL0.v, bH0.v, acc00);
      acc01 = wmma_bf(aL0.v, bH1.v, acc01);
      acc10 = wmma_bf(aL1.v, bH0.v, acc10);
      acc11 = wmma_bf(aL1.v, bH1.v, acc11);
    }

    wait_async0();     // next buffer filled
    __syncthreads();   // all waves done reading current buffer
  }

  const int rb = r0 + wave * 32 + 8 * lh;
  const int mA = m0 + l16, mB = m0 + 16 + l16;
#pragma unroll
  for (int v = 0; v < 8; ++v) {
    const int rA = rb + v;
    const int rB = rb + 16 + v;
    const float v00 = acc00[v], v01 = acc01[v], v10 = acc10[v], v11 = acc11[v];
    Out[(size_t)rA * NN + mA] = v00;
    Out[(size_t)rA * NN + mB] = v01;
    Out[(size_t)rB * NN + mA] = v10;
    Out[(size_t)rB * NN + mB] = v11;
    if (OutH) {
      auto wsplit = [&](size_t idx, float val) {
        const uint32_t hb = bf16h(val);
        OutH[idx] = (uint16_t)hb;
        OutL[idx] = (uint16_t)bf16h(val - bf16f(hb));
      };
      wsplit((size_t)rA * NN + mA, v00);
      wsplit((size_t)rA * NN + mB, v01);
      wsplit((size_t)rB * NN + mA, v10);
      wsplit((size_t)rB * NN + mB, v11);
    }
  }
}

// ---------------------------------------------------------------------------
// f32 diffusion (fallback path): Out = Z * adj^T with v_wmma_f32_16x16x4_f32
// grid = (N/32, R/64), block 128.
// ---------------------------------------------------------------------------
__global__ __launch_bounds__(128)
void diffuse_kernel(const float* __restrict__ Z, const float* __restrict__ A,
                    float* __restrict__ Out) {
  __shared__ float ldsZ[64 * LDP];
  __shared__ float ldsA[32 * LDP];

  const int tid  = threadIdx.x;
  const int wave = tid >> 5;
  const int lane = tid & 31;
  const int l16  = lane & 15;
  const int hi   = lane >> 4;
  const int m0   = blockIdx.x * 32;
  const int r0   = blockIdx.y * 64;

  v8f acc0 = {}; v8f acc1 = {};

  const int zr = tid >> 1;
  const int zo = (tid & 1) * 16;
  const int ar = tid >> 2;
  const int ao = (tid & 3) * 8;

  const float* Zrow = Z + (size_t)(r0 + zr) * NN + zo;
  const float* Arow = A + (size_t)(m0 + ar) * NN + ao;

  for (int k0 = 0; k0 < NN; k0 += 32) {
#pragma unroll
    for (int i = 0; i < 4; ++i)
      *(float4*)(&ldsZ[zr * LDP + zo + 4 * i]) = *(const float4*)(Zrow + k0 + 4 * i);
#pragma unroll
    for (int i = 0; i < 2; ++i)
      *(float4*)(&ldsA[ar * LDP + ao + 4 * i]) = *(const float4*)(Arow + k0 + 4 * i);
    __syncthreads();

    const float* za  = &ldsZ[(wave * 16 + l16) * LDP];
    const float* b0p = &ldsA[l16 * LDP];
    const float* b1p = &ldsA[(16 + l16) * LDP];
#pragma unroll
    for (int kk = 0; kk < 32; kk += 4) {
      const int ka = kk + 2 * hi;
      v2f a;  a.x  = za[ka];  a.y  = za[ka + 1];
      v2f b0; b0.x = b0p[ka]; b0.y = b0p[ka + 1];
      v2f b1; b1.x = b1p[ka]; b1.y = b1p[ka + 1];
      acc0 = wmma4(a, b0, acc0);
      acc1 = wmma4(a, b1, acc1);
    }
    __syncthreads();
  }

  const int rbase = r0 + wave * 16 + 8 * hi;
#pragma unroll
  for (int v = 0; v < 8; ++v) {
    Out[(size_t)(rbase + v) * NN + m0 + l16]      = acc0[v];
    Out[(size_t)(rbase + v) * NN + m0 + 16 + l16] = acc1[v];
  }
}

// ---------------------------------------------------------------------------
// gates: [r;u](128) = [Wf;Wu](128x288) * feats(288x32); writes U, RH (+hi/lo)
// ---------------------------------------------------------------------------
__global__ __launch_bounds__(128)
void gates_kernel(const float* __restrict__ Zg, const float* __restrict__ Z1,
                  const float* __restrict__ Z2,
                  const float* __restrict__ Wf, const float* __restrict__ bf,
                  const float* __restrict__ Wu, const float* __restrict__ bu,
                  const float* __restrict__ h,
                  float* __restrict__ RH, float* __restrict__ U,
                  uint16_t* __restrict__ RHH, uint16_t* __restrict__ RHL) {
  __shared__ float ldsW[128 * LDP];
  __shared__ float ldsF[32 * LDP];

  const int tid  = threadIdx.x;
  const int wave = tid >> 5;
  const int lane = tid & 31;
  const int l16  = lane & 15;
  const int hi   = lane >> 4;
  const int n0   = blockIdx.x * 32;
  const int b    = blockIdx.y;

  v8f acc00 = {}, acc01 = {}, acc10 = {}, acc11 = {};

  const float* wrow = (tid < 64) ? (Wf + tid * 288) : (Wu + (tid - 64) * 288);
  const int fr = tid >> 2;
  const int fo = (tid & 3) * 8;

  for (int c = 0; c < 9; ++c) {
    const int k0 = c * 32;
#pragma unroll
    for (int i = 0; i < 8; ++i)
      *(float4*)(&ldsW[tid * LDP + 4 * i]) = *(const float4*)(wrow + k0 + 4 * i);

    const float* src; int roff;
    if (k0 < 96)       { src = Zg; roff = k0;       }
    else if (k0 < 192) { src = Z1; roff = k0 - 96;  }
    else               { src = Z2; roff = k0 - 192; }
    const float* frow = src + (size_t)(b * 96 + roff + fr) * NN + n0 + fo;
#pragma unroll
    for (int i = 0; i < 2; ++i)
      *(float4*)(&ldsF[fr * LDP + fo + 4 * i]) = *(const float4*)(frow + 4 * i);
    __syncthreads();

    const float* a0p = &ldsW[(wave * 32 + l16) * LDP];
    const float* a1p = &ldsW[(wave * 32 + 16 + l16) * LDP];
#pragma unroll
    for (int kk = 0; kk < 32; kk += 4) {
      const int ka = kk + 2 * hi;
      v2f a0; a0.x = a0p[ka]; a0.y = a0p[ka + 1];
      v2f a1; a1.x = a1p[ka]; a1.y = a1p[ka + 1];
      v2f b0; b0.x = ldsF[ka * LDP + l16];      b0.y = ldsF[(ka + 1) * LDP + l16];
      v2f b1; b1.x = ldsF[ka * LDP + 16 + l16]; b1.y = ldsF[(ka + 1) * LDP + 16 + l16];
      acc00 = wmma4(a0, b0, acc00);
      acc01 = wmma4(a0, b1, acc01);
      acc10 = wmma4(a1, b0, acc10);
      acc11 = wmma4(a1, b1, acc11);
    }
    __syncthreads();
  }

  auto emit = [&](int m, int n, float val) {
    if (m < 64) {
      const float rv = sigmoidf_(val + bf[m]);
      const size_t idx = (size_t)(b * 64 + m) * NN + n;
      const float rh = rv * h[idx];
      RH[idx] = rh;
      if (RHH) {
        const uint32_t hb = bf16h(rh);
        RHH[idx] = (uint16_t)hb;
        RHL[idx] = (uint16_t)bf16h(rh - bf16f(hb));
      }
    } else {
      const int mu = m - 64;
      const float uv = sigmoidf_(val + bu[mu]);
      U[(size_t)(b * 64 + mu) * NN + n] = uv;
    }
  };

  const int nA = n0 + l16, nB = n0 + 16 + l16;
#pragma unroll
  for (int v = 0; v < 8; ++v) {
    const int m = wave * 32 + v + 8 * hi;
    emit(m,      nA, acc00[v]);
    emit(m,      nB, acc01[v]);
    emit(m + 16, nA, acc10[v]);
    emit(m + 16, nB, acc11[v]);
  }
}

// ---------------------------------------------------------------------------
// cgate: c = tanh(Wc * feats_c + bc); out = u*h + (1-u)*c
// ---------------------------------------------------------------------------
__global__ __launch_bounds__(128)
void cgate_kernel(const float* __restrict__ x,  const float* __restrict__ RH,
                  const float* __restrict__ Z1, const float* __restrict__ W1,
                  const float* __restrict__ Z2, const float* __restrict__ W2,
                  const float* __restrict__ Wc, const float* __restrict__ bc,
                  const float* __restrict__ U,  const float* __restrict__ h,
                  float* __restrict__ out) {
  __shared__ float ldsW[64 * LDP];
  __shared__ float ldsF[32 * LDP];

  const int tid  = threadIdx.x;
  const int wave = tid >> 5;
  const int lane = tid & 31;
  const int l16  = lane & 15;
  const int hi   = lane >> 4;
  const int n0   = blockIdx.x * 32;
  const int b    = blockIdx.y;

  v8f acc0 = {}, acc1 = {};

  const float* wrow = Wc + (tid >> 1) * 288;
  const int wo = (tid & 1) * 16;
  const int fr = tid >> 2;
  const int fo = (tid & 3) * 8;

  for (int c = 0; c < 9; ++c) {
    const int k0 = c * 32;
#pragma unroll
    for (int i = 0; i < 4; ++i)
      *(float4*)(&ldsW[(tid >> 1) * LDP + wo + 4 * i]) =
          *(const float4*)(wrow + k0 + wo + 4 * i);

    const float* src; int rb, roff;
    switch (c) {
      case 0:         src = x;  rb = b * 32; roff = 0;        break;
      case 1: case 2: src = RH; rb = b * 64; roff = k0 - 32;  break;
      case 3:         src = Z1; rb = b * 96; roff = 0;        break;
      case 4: case 5: src = W1; rb = b * 64; roff = k0 - 128; break;
      case 6:         src = Z2; rb = b * 96; roff = 0;        break;
      default:        src = W2; rb = b * 64; roff = k0 - 224; break;
    }
    const float* frow = src + (size_t)(rb + roff + fr) * NN + n0 + fo;
#pragma unroll
    for (int i = 0; i < 2; ++i)
      *(float4*)(&ldsF[fr * LDP + fo + 4 * i]) = *(const float4*)(frow + 4 * i);
    __syncthreads();

    const float* ap = &ldsW[(wave * 16 + l16) * LDP];
#pragma unroll
    for (int kk = 0; kk < 32; kk += 4) {
      const int ka = kk + 2 * hi;
      v2f a;  a.x  = ap[ka];                    a.y  = ap[ka + 1];
      v2f b0; b0.x = ldsF[ka * LDP + l16];      b0.y = ldsF[(ka + 1) * LDP + l16];
      v2f b1; b1.x = ldsF[ka * LDP + 16 + l16]; b1.y = ldsF[(ka + 1) * LDP + 16 + l16];
      acc0 = wmma4(a, b0, acc0);
      acc1 = wmma4(a, b1, acc1);
    }
    __syncthreads();
  }

#pragma unroll
  for (int v = 0; v < 8; ++v) {
    const int m = wave * 16 + v + 8 * hi;
    const float bias = bc[m];
    {
      const int n = n0 + l16;
      const float cv = tanhf(acc0[v] + bias);
      const size_t idx = (size_t)(b * 64 + m) * NN + n;
      const float uv = U[idx];
      out[idx] = uv * h[idx] + (1.0f - uv) * cv;
    }
    {
      const int n = n0 + 16 + l16;
      const float cv = tanhf(acc1[v] + bias);
      const size_t idx = (size_t)(b * 64 + m) * NN + n;
      const float uv = U[idx];
      out[idx] = uv * h[idx] + (1.0f - uv) * cv;
    }
  }
}

// ---------------------------------------------------------------------------
extern "C" void kernel_launch(void* const* d_in, const int* in_sizes, int n_in,
                              void* d_out, int out_size, void* d_ws, size_t ws_size,
                              hipStream_t stream) {
  (void)in_sizes; (void)n_in; (void)out_size;
  const float* x   = (const float*)d_in[0];
  const float* h   = (const float*)d_in[1];
  const float* adj = (const float*)d_in[2];
  const float* Wf  = (const float*)d_in[3];
  const float* bf  = (const float*)d_in[4];
  const float* Wu  = (const float*)d_in[5];
  const float* bu  = (const float*)d_in[6];
  const float* Wc  = (const float*)d_in[7];
  const float* bc  = (const float*)d_in[8];
  float* out = (float*)d_out;

  float* ws = (float*)d_ws;
  const size_t SZ96 = (size_t)16 * 96 * NN;   // 6,144,000
  const size_t SZ64 = (size_t)16 * 64 * NN;   // 4,096,000
  const size_t SZA  = (size_t)NN * NN;        // 16,000,000

  // f32 panels
  float* Zg = ws;
  float* Z1 = Zg + SZ96;
  float* Z2 = Z1 + SZ96;
  float* RH = Z2 + SZ96;
  float* W1 = RH + SZ64;
  float* W2 = W1 + SZ64;
  float* U  = W2 + SZ64;
  const size_t base_floats = 3 * SZ96 + 4 * SZ64;   // 34,816,000

  // bf16 hi/lo panels (split path)
  uint16_t* AH  = (uint16_t*)(ws + base_floats);
  uint16_t* AL  = AH  + SZA;
  uint16_t* ZgH = AL  + SZA;
  uint16_t* ZgL = ZgH + SZ96;
  uint16_t* Z1H = ZgL + SZ96;
  uint16_t* Z1L = Z1H + SZ96;
  uint16_t* RHH = Z1L + SZ96;
  uint16_t* RHL = RHH + SZ64;
  uint16_t* W1H = RHL + SZ64;
  uint16_t* W1L = W1H + SZ64;

  const size_t bf16_elems = 2 * SZA + 4 * SZ96 + 4 * SZ64;
  const size_t need_split = base_floats * 4 + bf16_elems * 2;  // ~285 MB
  const bool split = (ws_size >= need_split);

  if (split) {
    pack_kernel<<<6000, 256, 0, stream>>>(x, h, Zg, ZgH, ZgL);
    adjsplit_kernel<<<(int)((SZA / 8 + 255) / 256), 256, 0, stream>>>(adj, AH, AL);
    diffuse_bf16_kernel<<<dim3(NN / 32, 1536 / 128), 128, 0, stream>>>(
        ZgH, ZgL, AH, AL, Z1, Z1H, Z1L);
    diffuse_bf16_kernel<<<dim3(NN / 32, 1536 / 128), 128, 0, stream>>>(
        Z1H, Z1L, AH, AL, Z2, nullptr, nullptr);
    gates_kernel<<<dim3(NN / 32, 16), 128, 0, stream>>>(Zg, Z1, Z2, Wf, bf, Wu,
                                                        bu, h, RH, U, RHH, RHL);
    diffuse_bf16_kernel<<<dim3(NN / 32, 1024 / 128), 128, 0, stream>>>(
        RHH, RHL, AH, AL, W1, W1H, W1L);
    diffuse_bf16_kernel<<<dim3(NN / 32, 1024 / 128), 128, 0, stream>>>(
        W1H, W1L, AH, AL, W2, nullptr, nullptr);
    cgate_kernel<<<dim3(NN / 32, 16), 128, 0, stream>>>(x, RH, Z1, W1, Z2, W2,
                                                        Wc, bc, U, h, out);
  } else {
    pack_kernel<<<6000, 256, 0, stream>>>(x, h, Zg, nullptr, nullptr);
    diffuse_kernel<<<dim3(NN / 32, 1536 / 64), 128, 0, stream>>>(Zg, adj, Z1);
    diffuse_kernel<<<dim3(NN / 32, 1536 / 64), 128, 0, stream>>>(Z1, adj, Z2);
    gates_kernel<<<dim3(NN / 32, 16), 128, 0, stream>>>(Zg, Z1, Z2, Wf, bf, Wu,
                                                        bu, h, RH, U, nullptr,
                                                        nullptr);
    diffuse_kernel<<<dim3(NN / 32, 1024 / 64), 128, 0, stream>>>(RH, adj, W1);
    diffuse_kernel<<<dim3(NN / 32, 1024 / 64), 128, 0, stream>>>(W1, adj, W2);
    cgate_kernel<<<dim3(NN / 32, 16), 128, 0, stream>>>(x, RH, Z1, W1, Z2, W2,
                                                        Wc, bc, U, h, out);
  }
}
